// GroupQueryAttention_90898687852792
// MI455X (gfx1250) — matrix operand order repaired and checked by reading, not compiled, not run on hardware
//
#include <hip/hip_runtime.h>

// ---------------------------------------------------------------------------
// GQA (B=2, S=2048, H=2048, NH=32, G=8, D=64) for MI455X / gfx1250 (wave32)
// Matmuls: v_wmma_f32_16x16x32_bf16 (f32 accumulate).
// Tile staging: Tensor Data Mover (tensor_load_to_lds + s_wait_tensorcnt).
// ---------------------------------------------------------------------------

typedef unsigned short U16;
typedef unsigned int   U32;
typedef __attribute__((ext_vector_type(16))) __bf16 v16bf;
typedef __attribute__((ext_vector_type(8)))  float  v8f;
typedef __attribute__((ext_vector_type(4)))  unsigned int U32x4;
typedef __attribute__((ext_vector_type(2)))  unsigned int U32x2;
typedef __attribute__((ext_vector_type(4)))  float F32x4;
typedef __attribute__((ext_vector_type(4)))  unsigned int UV4;
typedef __attribute__((ext_vector_type(8)))  int IV8;
typedef __attribute__((ext_vector_type(4)))  int IV4;

struct BF16Frag { U32x4 lo, hi; };

static __device__ __forceinline__ U16 f2bf(float f) {
  U32 u = __float_as_uint(f);
  U32 r = u + 0x7FFFu + ((u >> 16) & 1u);   // round-to-nearest-even
  return (U16)(r >> 16);
}

// A-operand fragment (16x32, MxK): per half-wave h, K pairs live at
// elem offsets [h*8, h*8+8) and [16+h*8, 16+h*8+8)  (ISA 7.12.2 table).
static __device__ __forceinline__ v16bf frag_ld_a(const U16* p, int h) {
  BF16Frag f;
  f.lo = *reinterpret_cast<const U32x4*>(p + h * 8);
  f.hi = *reinterpret_cast<const U32x4*>(p + h * 8 + 16);
  return __builtin_bit_cast(v16bf, f);
}
// B-operand fragment (32x16, KxN, fed as [n][k] rows): per half-wave h,
// K pairs live at elem offsets [h*8, h*8+8) and [16+h*8, 16+h*8+8) (same as A).
static __device__ __forceinline__ v16bf frag_ld_b(const U16* p, int h) {
  BF16Frag f;
  f.lo = *reinterpret_cast<const U32x4*>(p + h * 8);
  f.hi = *reinterpret_cast<const U32x4*>(p + h * 8 + 16);
  return __builtin_bit_cast(v16bf, f);
}

// ---------------------------------------------------------------------------
// Tensor Data Mover: 2D bf16 tile load, global -> LDS, with LDS row padding.
// D# group0: count=1 | lds_addr | 57b global addr | type=2 ("image").
// D# group1: data_size=2B, pad cfg, tensor_dim0/1, tile_dim0/1, dim0 stride.
// pad_int_code: dwords-per-row before pad (0=2,1=4,2=8,3=16,4=32,...)
// pad_amt_code: pad dwords minus 1 (3 -> 4 dwords = 16B).
// ---------------------------------------------------------------------------
static __device__ __forceinline__ void tdm_load_2d(
    unsigned lds_addr, const void* gptr,
    unsigned tile_d0 /*elems*/, unsigned tile_d1 /*rows*/,
    unsigned tensor_d0 /*elems = row stride*/, unsigned tensor_d1 /*rows*/,
    unsigned pad_int_code, unsigned pad_amt_code) {
  unsigned long long ga = (unsigned long long)(size_t)gptr;
  UV4 g0;
  g0[0] = 1u;                                          // count=1, user D#
  g0[1] = lds_addr;                                    // LDS byte address
  g0[2] = (unsigned)ga;                                // global addr [31:0]
  g0[3] = ((unsigned)(ga >> 32) & 0x01FFFFFFu) | (2u << 30);  // [56:32]|type=2
  IV8 g1;
  g1[0] = (int)((1u << 16) |                           // data_size = 2 bytes
                (1u << 20) |                           // pad_enable
                (pad_int_code << 22) | (pad_amt_code << 25));
  g1[1] = (int)((tensor_d0 & 0xFFFFu) << 16);          // tensor_dim0 lo16
  g1[2] = (int)((tensor_d0 >> 16) | ((tensor_d1 & 0xFFFFu) << 16));
  g1[3] = (int)((tensor_d1 >> 16) | (tile_d0 << 16));  // dim1 hi | tile_dim0
  g1[4] = (int)tile_d1;                                // tile_dim1, tile_dim2=0
  g1[5] = (int)tensor_d0;                              // dim0 stride lo32
  g1[6] = 0;
  g1[7] = 0;
  IV4 z4 = {0, 0, 0, 0};
#if defined(__clang_major__) && (__clang_major__ >= 23)
  IV8 z8 = {0, 0, 0, 0, 0, 0, 0, 0};
  __builtin_amdgcn_tensor_load_to_lds(g0, g1, z4, z4, z8, 0);
#else
  __builtin_amdgcn_tensor_load_to_lds(g0, g1, z4, z4, 0);
#endif
}

// ---------------------------------------------------------------------------
// f32 -> bf16 elementwise convert (vectorized 4-wide)
// ---------------------------------------------------------------------------
__global__ __launch_bounds__(256) void cvt_f32_bf16(const float* __restrict__ in,
                                                    U16* __restrict__ out, int n4) {
  int i = blockIdx.x * 256 + threadIdx.x;
  if (i >= n4) return;
  F32x4 f = reinterpret_cast<const F32x4*>(in)[i];
  U32x2 o;
  o.x = (U32)f2bf(f.x) | ((U32)f2bf(f.y) << 16);
  o.y = (U32)f2bf(f.z) | ((U32)f2bf(f.w) << 16);
  reinterpret_cast<U32x2*>(out)[i] = o;
}

// ---------------------------------------------------------------------------
// W [K,N] f32  ->  Wt [N,K] bf16   (32x32 LDS-tiled transpose)
// ---------------------------------------------------------------------------
__global__ __launch_bounds__(256) void transpose_cvt(const float* __restrict__ in,
                                                     U16* __restrict__ out,
                                                     int K, int N) {
  __shared__ float t[32][33];
  const int x = threadIdx.x & 31, y = threadIdx.x >> 5;
  const int bx = blockIdx.x * 32;   // N direction
  const int by = blockIdx.y * 32;   // K direction
#pragma unroll
  for (int i = 0; i < 4; ++i)
    t[y + i * 8][x] = in[(size_t)(by + y + i * 8) * N + bx + x];
  __syncthreads();
#pragma unroll
  for (int i = 0; i < 4; ++i)
    out[(size_t)(bx + y + i * 8) * K + by + x] = f2bf(t[x][y + i * 8]);
}

// ---------------------------------------------------------------------------
// C[M,N] = A[M,K]*Bt[N,K]^T + bias   (bf16 in, f32 accumulate)
// 128x128 tile, BK=32, 8 waves (4x2), each wave 32x64 (8 WMMA tiles).
// Tiles staged by TDM (wave0 issues, TENSORcnt-tracked), double-buffered,
// one barrier per K-step, DMA fully overlapped with WMMA.
// ---------------------------------------------------------------------------
template <int OUT_F32>
__global__ __launch_bounds__(256) void gemm_bf16_wmma(
    const U16* __restrict__ A, const U16* __restrict__ Bt,
    const float* __restrict__ bias, void* __restrict__ Cv,
    int M, int N, int K) {
  __shared__ __align__(16) U16 As[2][128][40];   // 80B row stride (TDM-padded)
  __shared__ __align__(16) U16 Bs[2][128][40];

  const int tid = threadIdx.x, lane = tid & 31, wid = tid >> 5;
  const int wm = wid & 3, wn = wid >> 2;
  const int h = lane >> 4, r16 = lane & 15;
  const int mbase = blockIdx.y * 128;
  const int nbase = blockIdx.x * 128;

  const U16* Atile = A  + (size_t)mbase * K;
  const U16* Btile = Bt + (size_t)nbase * K;

  const v8f vzero = {0.f, 0.f, 0.f, 0.f, 0.f, 0.f, 0.f, 0.f};
  v8f acc[2][4];
#pragma unroll
  for (int i = 0; i < 2; ++i)
#pragma unroll
    for (int j = 0; j < 4; ++j) acc[i][j] = vzero;

  // prologue: DMA K-step 0 into buffer 0
  if (wid == 0) {
    tdm_load_2d((unsigned)(size_t)&As[0][0][0], Atile, 32, 128,
                (unsigned)K, (unsigned)M, 3, 3);
    tdm_load_2d((unsigned)(size_t)&Bs[0][0][0], Btile, 32, 128,
                (unsigned)K, (unsigned)N, 3, 3);
    __builtin_amdgcn_s_wait_tensorcnt(0);
  }
  __syncthreads();

  const int nk = K >> 5;
  for (int kk = 0; kk < nk; ++kk) {
    const int cur = kk & 1;
    // prefetch next K-step via TDM; overwritten buffer was last read in step
    // kk-1, whose trailing barrier all waves have already passed.
    if (wid == 0 && kk + 1 < nk) {
      const int nxt = cur ^ 1;
      tdm_load_2d((unsigned)(size_t)&As[nxt][0][0],
                  Atile + (size_t)(kk + 1) * 32, 32, 128,
                  (unsigned)K, (unsigned)M, 3, 3);
      tdm_load_2d((unsigned)(size_t)&Bs[nxt][0][0],
                  Btile + (size_t)(kk + 1) * 32, 32, 128,
                  (unsigned)K, (unsigned)N, 3, 3);
    }

    v16bf af[2], bfr[4];
#pragma unroll
    for (int i = 0; i < 2; ++i)
      af[i] = frag_ld_a(&As[cur][wm * 32 + i * 16 + r16][0], h);
#pragma unroll
    for (int j = 0; j < 4; ++j)
      bfr[j] = frag_ld_b(&Bs[cur][wn * 64 + j * 16 + r16][0], h);

#pragma unroll
    for (int i = 0; i < 2; ++i)
#pragma unroll
      for (int j = 0; j < 4; ++j)
        acc[i][j] = __builtin_amdgcn_wmma_f32_16x16x32_bf16(
            false, af[i], false, bfr[j], (short)0, acc[i][j], false, false);

    if (wid == 0) __builtin_amdgcn_s_wait_tensorcnt(0);  // next buffer landed
    __syncthreads();
  }

  // epilogue: C layout -> n = lane%16, m = r + 8*(lane/16)
#pragma unroll
  for (int j = 0; j < 4; ++j) {
    const int col = nbase + wn * 64 + j * 16 + r16;
    const float bval = bias[col];
#pragma unroll
    for (int i = 0; i < 2; ++i) {
#pragma unroll
      for (int r = 0; r < 8; ++r) {
        const int row = mbase + wm * 32 + i * 16 + r + h * 8;
        float v = acc[i][j][r] + bval;
        if (OUT_F32)
          ((float*)Cv)[(size_t)row * N + col] = v;
        else
          ((U16*)Cv)[(size_t)row * N + col] = f2bf(v);
      }
    }
  }
}

// ---------------------------------------------------------------------------
// Flash-style GQA attention.
// One workgroup per (b, head, 128-row q block); 8 waves, 16 q-rows each.
// K tiles: TDM double-buffered prefetch (one in flight, s_wait_tensorcnt 1).
// V tiles: cooperative transposed store (TDM cannot transpose).
// ---------------------------------------------------------------------------
__global__ __launch_bounds__(256) void gqa_attention(
    const U16* __restrict__ Q,   // [B*S, 2048] bf16
    const U16* __restrict__ Kg,  // [B*S, 512]  bf16
    const U16* __restrict__ Vg,  // [B*S, 512]  bf16
    U16* __restrict__ O) {       // [B*S, 2048] bf16 (head-major merge order)
  __shared__ __align__(16) U16 Qs[128][72];
  __shared__ __align__(16) U16 Ks[2][64][72];   // 144B row stride (TDM-padded)
  __shared__ __align__(16) U16 Vs[64][72];      // transposed: [d][kpos]
  __shared__ __align__(16) U16 Ps[8][16][72];   // per-wave P staging

  const int tid = threadIdx.x, lane = tid & 31, wid = tid >> 5;
  const int h = lane >> 4, r16 = lane & 15;

  const int qblk = blockIdx.x & 15;
  const int head = (blockIdx.x >> 4) & 31;
  const int b    = blockIdx.x >> 9;
  const int g    = head >> 2;          // R = 4 q-heads per kv group

  const size_t qrow0 = (size_t)b * 2048 + qblk * 128;
  const size_t krow0 = (size_t)b * 2048;

  // kick off K block 0 DMA immediately
  if (wid == 0)
    tdm_load_2d((unsigned)(size_t)&Ks[0][0][0],
                Kg + krow0 * 512 + g * 64, 64, 64, 512, 4096, 4, 3);

  // load Q tile [128 x 64] once
  {
    const int row = tid >> 1, half = tid & 1;
    const U16* src = Q + (qrow0 + row) * 2048 + head * 64 + half * 32;
#pragma unroll
    for (int o = 0; o < 4; ++o)
      *reinterpret_cast<U32x4*>(&Qs[row][half * 32 + o * 8]) =
          *reinterpret_cast<const U32x4*>(src + o * 8);
  }

  const v8f vzero = {0.f, 0.f, 0.f, 0.f, 0.f, 0.f, 0.f, 0.f};
  float mrow[8], lrow[8];
  v8f oacc[4];
#pragma unroll
  for (int r = 0; r < 8; ++r) { mrow[r] = -1e30f; lrow[r] = 0.f; }
#pragma unroll
  for (int j = 0; j < 4; ++j) oacc[j] = vzero;

  __syncthreads();

  for (int blk = 0; blk < 32; ++blk) {
    const int cur = blk & 1;
    // stage V^T [64x64] cooperatively (prev compute done: trailing barrier)
    {
      const int row = tid >> 2, cg = tid & 3;
      const U16* vsrc = Vg + (krow0 + blk * 64 + row) * 512 + g * 64 + cg * 16;
      union { U32x4 q; U16 s[8]; } u0, u1;
      u0.q = *reinterpret_cast<const U32x4*>(vsrc);
      u1.q = *reinterpret_cast<const U32x4*>(vsrc + 8);
#pragma unroll
      for (int e = 0; e < 8; ++e) {
        Vs[cg * 16 + e][row]     = u0.s[e];
        Vs[cg * 16 + 8 + e][row] = u1.s[e];
      }
    }
    // prefetch next K block; wait for current (TDM retires in order)
    if (wid == 0) {
      if (blk + 1 < 32) {
        tdm_load_2d((unsigned)(size_t)&Ks[cur ^ 1][0][0],
                    Kg + (krow0 + (size_t)(blk + 1) * 64) * 512 + g * 64,
                    64, 64, 512, 4096, 4, 3);
        __builtin_amdgcn_s_wait_tensorcnt(1);
      } else {
        __builtin_amdgcn_s_wait_tensorcnt(0);
      }
    }
    __syncthreads();

    // S = Q * K^T  (this wave's 16 rows x 64 kpos)
    v8f sacc[4];
#pragma unroll
    for (int j = 0; j < 4; ++j) sacc[j] = vzero;
#pragma unroll
    for (int ks = 0; ks < 2; ++ks) {
      v16bf af = frag_ld_a(&Qs[wid * 16 + r16][ks * 32], h);
#pragma unroll
      for (int j = 0; j < 4; ++j) {
        v16bf bfr = frag_ld_b(&Ks[cur][j * 16 + r16][ks * 32], h);
        sacc[j] = __builtin_amdgcn_wmma_f32_16x16x32_bf16(
            false, af, false, bfr, (short)0, sacc[j], false, false);
      }
    }

    // online softmax over this k-block; C layout: row = r + 8h, col = r16
#pragma unroll
    for (int r = 0; r < 8; ++r) {
      float s0 = sacc[0][r] * 0.125f, s1 = sacc[1][r] * 0.125f;
      float s2 = sacc[2][r] * 0.125f, s3 = sacc[3][r] * 0.125f;
      float mx = fmaxf(fmaxf(s0, s1), fmaxf(s2, s3));
#pragma unroll
      for (int m = 1; m < 16; m <<= 1) mx = fmaxf(mx, __shfl_xor(mx, m, 32));
      float mnew  = fmaxf(mrow[r], mx);
      float alpha = __expf(mrow[r] - mnew);
      float p0 = __expf(s0 - mnew), p1 = __expf(s1 - mnew);
      float p2 = __expf(s2 - mnew), p3 = __expf(s3 - mnew);
      float rs = p0 + p1 + p2 + p3;
#pragma unroll
      for (int m = 1; m < 16; m <<= 1) rs += __shfl_xor(rs, m, 32);
      lrow[r] = lrow[r] * alpha + rs;
      mrow[r] = mnew;
#pragma unroll
      for (int j = 0; j < 4; ++j) oacc[j][r] *= alpha;
      Ps[wid][r + h * 8][r16]      = f2bf(p0);
      Ps[wid][r + h * 8][16 + r16] = f2bf(p1);
      Ps[wid][r + h * 8][32 + r16] = f2bf(p2);
      Ps[wid][r + h * 8][48 + r16] = f2bf(p3);
    }

    // O += P * V   (A = P [16 x 64kpos], B = Vs rows are [d][kpos])
#pragma unroll
    for (int ks = 0; ks < 2; ++ks) {
      v16bf af = frag_ld_a(&Ps[wid][r16][ks * 32], h);
#pragma unroll
      for (int j = 0; j < 4; ++j) {
        v16bf bfr = frag_ld_b(&Vs[j * 16 + r16][ks * 32], h);
        oacc[j] = __builtin_amdgcn_wmma_f32_16x16x32_bf16(
            false, af, false, bfr, (short)0, oacc[j], false, false);
      }
    }
    __syncthreads();
  }

  // normalize and store (head-major merge: col = head*64 + d)
#pragma unroll
  for (int r = 0; r < 8; ++r) {
    float inv = 1.0f / lrow[r];
    const size_t orow = qrow0 + wid * 16 + r + h * 8;
#pragma unroll
    for (int j = 0; j < 4; ++j)
      O[orow * 2048 + head * 64 + j * 16 + r16] = f2bf(oacc[j][r] * inv);
  }
}

// ---------------------------------------------------------------------------
// Host launcher
// ---------------------------------------------------------------------------
extern "C" void kernel_launch(void* const* d_in, const int* in_sizes, int n_in,
                              void* d_out, int out_size, void* d_ws, size_t ws_size,
                              hipStream_t stream) {
  (void)in_sizes; (void)n_in; (void)out_size; (void)ws_size;
  const float* x  = (const float*)d_in[0];
  const float* Wq = (const float*)d_in[1];
  const float* bq = (const float*)d_in[2];
  const float* Wk = (const float*)d_in[3];
  const float* bk = (const float*)d_in[4];
  const float* Wv = (const float*)d_in[5];
  const float* bv = (const float*)d_in[6];
  const float* Wo = (const float*)d_in[7];
  const float* bo = (const float*)d_in[8];

  char* ws = (char*)d_ws;
  size_t off = 0;
  auto take = [&](size_t bytes) -> void* {
    void* p = (void*)(ws + off);
    off += (bytes + 255) & ~(size_t)255;
    return p;
  };
  // total ~76 MB of workspace
  U16* xb  = (U16*)take((size_t)4096 * 2048 * 2);
  U16* WqT = (U16*)take((size_t)2048 * 2048 * 2);
  U16* WkT = (U16*)take((size_t)512  * 2048 * 2);
  U16* WvT = (U16*)take((size_t)512  * 2048 * 2);
  U16* WoT = (U16*)take((size_t)2048 * 2048 * 2);
  U16* qb  = (U16*)take((size_t)4096 * 2048 * 2);
  U16* kb  = (U16*)take((size_t)4096 * 512  * 2);
  U16* vb  = (U16*)take((size_t)4096 * 512  * 2);
  U16* ab  = (U16*)take((size_t)4096 * 2048 * 2);

  // 1) precision conversion / weight transposition
  cvt_f32_bf16<<<(4096 * 2048 / 4) / 256, 256, 0, stream>>>(x, xb, 4096 * 2048 / 4);
  transpose_cvt<<<dim3(64, 64), 256, 0, stream>>>(Wq, WqT, 2048, 2048);
  transpose_cvt<<<dim3(16, 64), 256, 0, stream>>>(Wk, WkT, 2048, 512);
  transpose_cvt<<<dim3(16, 64), 256, 0, stream>>>(Wv, WvT, 2048, 512);
  transpose_cvt<<<dim3(64, 64), 256, 0, stream>>>(Wo, WoT, 2048, 2048);

  // 2) QKV projections (bf16 out)
  gemm_bf16_wmma<0><<<dim3(16, 32), 256, 0, stream>>>(xb, WqT, bq, qb, 4096, 2048, 2048);
  gemm_bf16_wmma<0><<<dim3(4, 32),  256, 0, stream>>>(xb, WkT, bk, kb, 4096, 512, 2048);
  gemm_bf16_wmma<0><<<dim3(4, 32),  256, 0, stream>>>(xb, WvT, bv, vb, 4096, 512, 2048);

  // 3) grouped attention (B*NH*S/128 = 1024 workgroups)
  gqa_attention<<<1024, 256, 0, stream>>>(qb, kb, vb, ab);

  // 4) output projection (f32 out)
  gemm_bf16_wmma<1><<<dim3(16, 32), 256, 0, stream>>>(ab, WoT, bo, d_out, 4096, 2048, 2048);
}
